// DeltaDNNGLM_5042291605669
// MI455X (gfx1250) — matrix operand order repaired and checked by reading, not compiled, not run on hardware
//
#include <hip/hip_runtime.h>

// ---------------------------------------------------------------------------
// DeltaDNN-GLM scan for MI455X (gfx1250, wave32, WMMA), transposed formulation.
//
// h^T = W^T @ act^T: batch is N, pre-packed transposed weights are A.
//   * B fragments (activations) = two 16B runs (k 8h.., 16+8h..) of row-major h[batch][k]
//   * D tiles write 8 contiguous bf16 per lane -> one packed ds_store_b128
//   * GEMM2: 4x4 outer-product register blocking, each fragment loaded once
// NEW: x is pre-converted to a padded bf16 layout xb[b][t][64] (prologue
// kernel), and the scan double-buffers z in LDS, prefetching step t+1's rows
// with GLOBAL_LOAD_ASYNC_TO_LDS_B128 during stage 5 and waiting on
// ASYNCcnt at the top of the loop (falls back to direct loads if ws is small).
// ---------------------------------------------------------------------------

#define Bsz   1024
#define Ssz   1000
#define Fsz   63
#define Hsz   512
#define OUTsz 2
#define MT    64                 // batch rows per workgroup
#define NWG   (Bsz / MT)         // 16 workgroups
#define ZRS   72                 // z LDS row stride (padded, bank-spread)
#define HRS   520                // h LDS row stride (padded, bank-spread)

typedef __bf16 bf16;
typedef __attribute__((ext_vector_type(16))) __bf16 v16bf;
typedef __attribute__((ext_vector_type(8)))  __bf16 v8bf;
typedef __attribute__((ext_vector_type(8)))  float  v8f;

// workspace layout (bf16 elements)
#define W1P_OFF 0                       // W1^T A-frags: 32 mt x 2 kt
#define W2P_OFF 32768                   // W2^T A-frags: 32 mt x 16 kt
#define W3P_OFF (32768 + 262144)        // W3^T A-frags: 1 mt (pad 16) x 16 kt
#define WPK_TOTAL (32768 + 262144 + 8192)
#define XB_OFF  WPK_TOTAL               // xb: [B][S][64] bf16 (optional)

// ---------------------------------------------------------------------------
// Prologue 1: pack f32 weights into bf16 WMMA *A*-fragments of W^T.
// A-frag (16x32 bf16), lane l (h = l>>4, m = l&15), element e:
//   e<8 : k = kt*32 + h*8 + e ;  e>=8: k = kt*32 + 16 + h*8 + (e-8)
// value = W^T[mtile*16+m][k] = W[k][mtile*16+m]. Lane data contiguous (32 B).
// ---------------------------------------------------------------------------
__global__ __launch_bounds__(256) void pack_weights(
    const float* __restrict__ W1, const float* __restrict__ W2,
    const float* __restrict__ W3, bf16* __restrict__ wsb)
{
  int id = blockIdx.x * blockDim.x + threadIdx.x;
  if (id < 32768) {                                  // W1^T: mt 0..31, kt 0..1
    int e = id & 15, l = (id >> 4) & 31, f = id >> 9;
    int kt = f & 1, mt = f >> 1;
    int h = l >> 4;
    int k = kt * 32 + ((e < 8) ? (h * 8 + e) : (16 + h * 8 + (e - 8)));
    int m = mt * 16 + (l & 15);
    wsb[W1P_OFF + id] = (bf16)W1[k * Hsz + m];
  } else if (id < 32768 + 262144) {                  // W2^T: mt 0..31, kt 0..15
    int id2 = id - 32768;
    int e = id2 & 15, l = (id2 >> 4) & 31, f = id2 >> 9;
    int kt = f & 15, mt = f >> 4;
    int h = l >> 4;
    int k = kt * 32 + ((e < 8) ? (h * 8 + e) : (16 + h * 8 + (e - 8)));
    int m = mt * 16 + (l & 15);
    wsb[W2P_OFF + id2] = (bf16)W2[k * Hsz + m];
  } else if (id < WPK_TOTAL) {                       // W3^T: kt 0..15 (M pad 16)
    int id3 = id - 32768 - 262144;
    int e = id3 & 15, l = (id3 >> 4) & 31, kt = id3 >> 9;
    int h = l >> 4;
    int k = kt * 32 + ((e < 8) ? (h * 8 + e) : (16 + h * 8 + (e - 8)));
    int m = l & 15;
    wsb[W3P_OFF + id3] = (m < OUTsz) ? (bf16)W3[k * OUTsz + m] : (bf16)0.0f;
  }
}

// ---------------------------------------------------------------------------
// Prologue 2: pre-convert x into padded bf16 rows xb[b][t][64]:
// col 0..1 = 0 (Wc slot, overwritten in-loop), col c>=2 = x[b][t][c-1].
// One z row becomes exactly 128 contiguous bytes -> pure async byte-copy.
// ---------------------------------------------------------------------------
__global__ __launch_bounds__(256) void convert_x(const float* __restrict__ x,
                                                 bf16* __restrict__ xb)
{
  size_t id = (size_t)blockIdx.x * 256 + threadIdx.x;   // over B*S*64
  if (id >= (size_t)Bsz * Ssz * 64) return;
  int    c  = (int)(id & 63);
  size_t bt = id >> 6;                                  // b*S + t
  float v = (c >= 2) ? x[bt * Fsz + (c - 1)] : 0.0f;
  xb[id] = (bf16)v;
}

// ---------------------------------------------------------------------------
// CDNA5 async-copy helpers (ASYNCcnt path, per ISA 10.x / 15.18.3)
// ---------------------------------------------------------------------------
__device__ __forceinline__ void async_copy_b128(const void* gsrc, void* ldst) {
  unsigned loff = (unsigned)(unsigned long long)ldst;   // LDS byte offset
  unsigned long long g = (unsigned long long)gsrc;
  asm volatile("global_load_async_to_lds_b128 %0, %1, off"
               :: "v"(loff), "v"(g) : "memory");
}
__device__ __forceinline__ void wait_async0() {
  asm volatile("s_wait_asynccnt 0x0" ::: "memory");
}

// Issue async copy of step t's 64 z-rows (128 B each) into zbuf.
// 512 b128 chunks over 256 threads -> 2 async ops per thread.
__device__ __forceinline__ void issue_x_async(const bf16* __restrict__ xb,
                                              bf16* zbuf, int b0, int t, int tid) {
#pragma unroll
  for (int pass = 0; pass < 2; ++pass) {
    int idx = tid + pass * 256;
    int row = idx >> 3, ch = idx & 7;
    const bf16* src = xb + ((size_t)(b0 + row) * Ssz + t) * 64 + ch * 8;
    async_copy_b128(src, zbuf + row * ZRS + ch * 8);
  }
}

// ---------------------------------------------------------------------------
// Fragment helpers
// ---------------------------------------------------------------------------
__device__ __forceinline__ v8f wmma_bf16(v16bf a, v16bf b, v8f c) {
  return __builtin_amdgcn_wmma_f32_16x16x32_bf16(false, a, false, b, (short)0,
                                                 c, false, false);
}

__device__ __forceinline__ v16bf load_frag_g(const bf16* packed, int f, int lane) {
  return *(const v16bf*)(packed + (size_t)f * 512 + lane * 16);
}

__device__ __forceinline__ v16bf load_b_lds(const bf16* buf, int RS, int ntG,
                                            int kt, int lane) {
  const bf16* p = buf + (ntG * 16 + (lane & 15)) * RS + kt * 32 + (lane >> 4) * 8;
  v8bf lo = *(const v8bf*)p;
  v8bf hi = *(const v8bf*)(p + 16);
  return __builtin_shufflevector(lo, hi, 0, 1, 2, 3, 4, 5, 6, 7,
                                 8, 9, 10, 11, 12, 13, 14, 15);
}

__device__ __forceinline__ void store_act(bf16* buf, int RS, int mtG, int ntG,
                                          v8f acc, v8f bias, int lane) {
  int n  = ntG * 16 + (lane & 15);
  int mo = mtG * 16 + ((lane >> 4) << 3);
  v8bf o;
#pragma unroll
  for (int r = 0; r < 8; ++r) {
    float v = acc[r] + bias[r];
    v = v > 0.0f ? v : 0.0f;
    o[r] = (bf16)v;
  }
  *(v8bf*)(buf + n * RS + mo) = o;
}

// ---------------------------------------------------------------------------
// Main persistent scan kernel: 16 WGs x 256 threads (8 waves).
// Wave w owns hidden M-tiles [w*4, w*4+4) and all 4 batch N-tiles.
// ---------------------------------------------------------------------------
__global__ __launch_bounds__(256) void delta_scan(
    const float* __restrict__ x, const float* __restrict__ W0,
    const float* __restrict__ scal, const float* __restrict__ b1g,
    const float* __restrict__ b2g, const float* __restrict__ b3g,
    const bf16* __restrict__ wsb, const bf16* __restrict__ xb, int useXb,
    float* __restrict__ out_sig, float* __restrict__ out_dw,
    float* __restrict__ out_W)
{
  __shared__ bf16  zA[2][MT * ZRS];   // double-buffered z, row-major [batch][64]
  __shared__ bf16  h1[MT * HRS];      // h1 row-major [batch][512]
  __shared__ bf16  h2[MT * HRS];      // h2 row-major [batch][512]
  __shared__ float Wst[MT * OUTsz];   // recurrent state
  __shared__ float b1s[Hsz];
  __shared__ float b2s[Hsz];

  const int tid  = threadIdx.x;
  const int wave = tid >> 5;
  const int lane = tid & 31;
  const int b0   = blockIdx.x * MT;

  const bf16* W1p = wsb + W1P_OFF;
  const bf16* W2p = wsb + W2P_OFF;
  const bf16* W3p = wsb + W3P_OFF;

  for (int i = tid; i < Hsz; i += 256) { b1s[i] = b1g[i]; b2s[i] = b2g[i]; }
  if (tid < MT * OUTsz) Wst[tid] = W0[(size_t)b0 * OUTsz + tid];
  __syncthreads();

  const int   wmt = wave * 4;
  const float sc0 = scal[0], sc1 = scal[1];
  const float b30 = b3g[0],  b31 = b3g[1];

  if (useXb) issue_x_async(xb, &zA[0][0], b0, 0, tid);   // prime t=0

#pragma unroll 1
  for (int t = 0; t < Ssz; ++t) {
    bf16* zcur = &zA[t & 1][0];

    // ---- stage 1: z = [Wc | x[:,t,1:]] (async-staged or direct) ----
    if (useXb) {
      wait_async0();                    // own wave's async copies complete
      __syncthreads();                  // all waves' copies visible
      if (tid < MT * OUTsz) zcur[(tid >> 1) * ZRS + (tid & 1)] = (bf16)Wst[tid];
    } else {
      for (int idx = tid; idx < MT * (Fsz - 1); idx += 256) {
        int r = idx / (Fsz - 1);
        int c = idx - r * (Fsz - 1);
        zcur[r * ZRS + 2 + c] =
            (bf16)x[(size_t)(b0 + r) * (Ssz * Fsz) + t * Fsz + 1 + c];
      }
      if (tid < MT * OUTsz) zcur[(tid >> 1) * ZRS + (tid & 1)] = (bf16)Wst[tid];
    }
    __syncthreads();

    // ---- stage 2: GEMM1  h1^T = W1^T @ z^T  (K=64) ----
    {
      v8f z8 = {};
      v8f acc[4][4];
#pragma unroll
      for (int i = 0; i < 4; ++i)
#pragma unroll
        for (int j = 0; j < 4; ++j) acc[i][j] = z8;
#pragma unroll
      for (int kt = 0; kt < 2; ++kt) {
        v16bf a[4], b[4];
#pragma unroll
        for (int i = 0; i < 4; ++i) a[i] = load_frag_g(W1p, (wmt + i) * 2 + kt, lane);
#pragma unroll
        for (int j = 0; j < 4; ++j) b[j] = load_b_lds(zcur, ZRS, j, kt, lane);
#pragma unroll
        for (int i = 0; i < 4; ++i)
#pragma unroll
          for (int j = 0; j < 4; ++j) acc[i][j] = wmma_bf16(a[i], b[j], acc[i][j]);
      }
#pragma unroll
      for (int i = 0; i < 4; ++i) {
        v8f bias = *(const v8f*)(b1s + (wmt + i) * 16 + ((lane >> 4) << 3));
#pragma unroll
        for (int j = 0; j < 4; ++j) store_act(h1, HRS, wmt + i, j, acc[i][j], bias, lane);
      }
    }
    __syncthreads();

    // ---- stage 3: GEMM2  h2^T = W2^T @ h1^T  (K=512, dominant) ----
    {
      v8f z8 = {};
      v8f acc[4][4];
#pragma unroll
      for (int i = 0; i < 4; ++i)
#pragma unroll
        for (int j = 0; j < 4; ++j) acc[i][j] = z8;
#pragma unroll 1
      for (int kt = 0; kt < 16; ++kt) {
        v16bf a[4], b[4];
#pragma unroll
        for (int i = 0; i < 4; ++i) a[i] = load_frag_g(W2p, (wmt + i) * 16 + kt, lane);
#pragma unroll
        for (int j = 0; j < 4; ++j) b[j] = load_b_lds(h1, HRS, j, kt, lane);
#pragma unroll
        for (int i = 0; i < 4; ++i)
#pragma unroll
          for (int j = 0; j < 4; ++j) acc[i][j] = wmma_bf16(a[i], b[j], acc[i][j]);
      }
#pragma unroll
      for (int i = 0; i < 4; ++i) {
        v8f bias = *(const v8f*)(b2s + (wmt + i) * 16 + ((lane >> 4) << 3));
#pragma unroll
        for (int j = 0; j < 4; ++j) store_act(h2, HRS, wmt + i, j, acc[i][j], bias, lane);
      }
    }
    __syncthreads();

    // ---- stage 4: GEMM3  dW^T = W3^T @ h2^T, state update ----
    if (wave < 4) {                      // nt = wave (16 batch rows each)
      v8f acc = {};
#pragma unroll 1
      for (int kt = 0; kt < 16; ++kt) {
        v16bf a = load_frag_g(W3p, kt, lane);
        v16bf b = load_b_lds(h2, HRS, wave, kt, lane);
        acc = wmma_bf16(a, b, acc);
      }
      if (lane < 16) {                   // reg r = out idx (0,1), batch = wave*16+lane
        int row = wave * 16 + lane;
#pragma unroll
        for (int r = 0; r < OUTsz; ++r) {
          float dw = (r == 0 ? sc0 : sc1) * (acc[r] + (r == 0 ? b30 : b31));
          float wn = Wst[row * OUTsz + r] + dw;
          Wst[row * OUTsz + r] = wn;
          if (t == Ssz - 1) out_dw[(size_t)(b0 + row) * OUTsz + r] = dw;
        }
      }
    }
    __syncthreads();

    // ---- stage 5: outputs + sigmoid head; prefetch/async-stage t+1 ----
    if (useXb) {
      if (t + 1 < Ssz) issue_x_async(xb, &zA[(t + 1) & 1][0], b0, t + 1, tid);
    }
    if (tid < MT) {
      int row = tid;
      const float* xrow = x + (size_t)(b0 + row) * (Ssz * Fsz) + t * Fsz;
      float w0v = Wst[row * OUTsz + 0];
      float w1v = Wst[row * OUTsz + 1];
      float wsum = w0v * xrow[0] + w1v;
      float sg = 1.0f / (1.0f + __builtin_expf(-wsum));
      size_t bt = (size_t)(b0 + row) * Ssz + t;
      out_sig[bt]       = sg;
      out_W[bt * 2 + 0] = w0v;
      out_W[bt * 2 + 1] = w1v;
      if (!useXb && t + 1 < Ssz) {       // fallback: warm WGP$/L2 for next step
        __builtin_prefetch(xrow + Fsz, 0, 0);
        __builtin_prefetch(xrow + Fsz + 32, 0, 0);
      }
    }
    __syncthreads();
  }
}

// ---------------------------------------------------------------------------
extern "C" void kernel_launch(void* const* d_in, const int* in_sizes, int n_in,
                              void* d_out, int out_size, void* d_ws, size_t ws_size,
                              hipStream_t stream)
{
  const float* x  = (const float*)d_in[0];
  const float* W  = (const float*)d_in[1];
  const float* sc = (const float*)d_in[2];
  const float* W1 = (const float*)d_in[3];
  const float* b1 = (const float*)d_in[4];
  const float* W2 = (const float*)d_in[5];
  const float* b2 = (const float*)d_in[6];
  const float* W3 = (const float*)d_in[7];
  const float* b3 = (const float*)d_in[8];

  float* out     = (float*)d_out;
  float* out_sig = out;                                           // (B,S,1)
  float* out_dw  = out + (size_t)Bsz * Ssz;                       // (B,2)
  float* out_W   = out + (size_t)Bsz * Ssz + (size_t)Bsz * OUTsz; // (B,S,2)

  bf16* wsb = (bf16*)d_ws;
  bf16* xb  = wsb + XB_OFF;

  size_t need_fast = ((size_t)WPK_TOTAL + (size_t)Bsz * Ssz * 64) * sizeof(bf16);
  int useXb = (ws_size >= need_fast) ? 1 : 0;

  pack_weights<<<(WPK_TOTAL + 255) / 256, 256, 0, stream>>>(W1, W2, W3, wsb);
  if (useXb) {
    size_t nconv = (size_t)Bsz * Ssz * 64;
    convert_x<<<(unsigned)((nconv + 255) / 256), 256, 0, stream>>>(x, xb);
  }
  delta_scan<<<NWG, 256, 0, stream>>>(x, W, sc, b1, b2, b3, wsb, xb, useXb,
                                      out_sig, out_dw, out_W);
}
